// HyperConv2d_50208167690695
// MI455X (gfx1250) — matrix operand order repaired and checked by reading, not compiled, not run on hardware
//
#include <hip/hip_runtime.h>
#include <hip/hip_bf16.h>
#include <stdint.h>

// ---------------------------------------------------------------------------
// HyperConv2d on gfx1250 (MI455X): implicit-GEMM 3x3 conv on the bf16 WMMA
// pipe with bf16x3 error compensation (hi/lo split, f32 accumulate):
//   x*w ~= hi(x)*hi(w) + hi(x)*lo(w) + lo(x)*hi(w)   (f32-faithful)
// Matrix op: V_WMMA_F32_16X16X32_BF16, D(16x16 f32) = A(16x32) x B(32x16) + C
// ---------------------------------------------------------------------------

typedef __attribute__((ext_vector_type(8)))  __bf16 v8bf;
typedef __attribute__((ext_vector_type(16))) __bf16 v16bf;
typedef __attribute__((ext_vector_type(8)))  float  v8f;

#define BATCH   8
#define CIN     64
#define COUT    64
#define KK      3
#define MID     64
#define NPARAM  4
#define NLAYER  5
#define HW      256
#define KDIM    (9 * CIN)          /* 576: k = tap*64 + ci              */
#define PCOLS   18                 /* 16 + 2 halo                       */
#define PSTR    72                 /* 64 ch + 8 pad (bank skew, bf16)   */

// ---------------- Embedder stage 1: w1[b][ci][m] (B x 4096, f32) -----------
__global__ __launch_bounds__(256) void hk_embed_w1(
    const float* __restrict__ h, const float* __restrict__ index,
    const float* __restrict__ W_wp, const float* __restrict__ b_wp,
    const float* __restrict__ W_wi, const float* __restrict__ b_wi,
    float* __restrict__ w1)
{
    int t = blockIdx.x * 256 + threadIdx.x;          // 0 .. 32767
    int b = t >> 12;
    int r = t & 4095;                                // ci*64 + m
    float acc = b_wp[r] + b_wi[r];
#pragma unroll
    for (int p = 0; p < NPARAM; ++p) acc += h[b * NPARAM + p] * W_wp[r * NPARAM + p];
#pragma unroll
    for (int l = 0; l < NLAYER; ++l) acc += index[l] * W_wi[r * NLAYER + l];
    w1[t] = acc;
}

// ---- Embedder stage 2: packed hi/lo bf16 weights Wt[b][o][k], k=tap*64+ci -
__global__ __launch_bounds__(256) void hk_embed_w2(
    const float* __restrict__ w1, const float* __restrict__ W_we,
    const float* __restrict__ b_we, const float* __restrict__ conv_w,
    __bf16* __restrict__ Wt_hi, __bf16* __restrict__ Wt_lo)
{
    int t = blockIdx.x * 256 + threadIdx.x;          // 0 .. 294911
    int k  = t % KDIM;
    int o  = (t / KDIM) & 63;
    int b  = t / (KDIM * COUT);
    int tap = k >> 6;                                // ky*3+kx
    int ci  = k & 63;
    int row = o * 9 + tap;                           // W_we row (Cout*9 x MID)
    const float* w1p = w1 + b * (CIN * MID) + ci * MID;
    const float* wep = W_we + row * MID;
    float acc = b_we[row];
#pragma unroll 4
    for (int m = 0; m < MID; ++m) acc += w1p[m] * wep[m];
    acc += conv_w[((o * CIN + ci) * KK + tap / 3) * KK + tap % 3];
    __bf16 hi = (__bf16)acc;
    __bf16 lo = (__bf16)(acc - (float)hi);
    Wt_hi[t] = hi;
    Wt_lo[t] = lo;
}

// ---------------- Embedder stage 3: bias[b][o] (f32) -----------------------
__global__ __launch_bounds__(256) void hk_embed_bias(
    const float* __restrict__ h, const float* __restrict__ index,
    const float* __restrict__ W_bp, const float* __restrict__ b_bp,
    const float* __restrict__ W_bi, const float* __restrict__ b_bi,
    const float* __restrict__ conv_b, float* __restrict__ bias)
{
    int t = blockIdx.x * 256 + threadIdx.x;          // 0 .. 511
    if (t >= BATCH * COUT) return;
    int b = t >> 6, o = t & 63;
    float acc = conv_b[o] + b_bp[o] + b_bi[o];
#pragma unroll
    for (int p = 0; p < NPARAM; ++p) acc += h[b * NPARAM + p] * W_bp[o * NPARAM + p];
#pragma unroll
    for (int l = 0; l < NLAYER; ++l) acc += index[l] * W_bi[o * NLAYER + l];
    bias[t] = acc;
}

// ---------------------------------------------------------------------------
// Conv kernel: one block = one (batch, 16x16 spatial tile), all 64 out-ch.
// 8 wave32s: wave w -> out-ch tile (w&3)*16, rows py = (w>>2)*8 .. +7.
// LDS: hi/lo patches [18 rows][18 cols][72 ch-padded] bf16 (2 x 46656 B).
// K loop: 9 taps x 2 half-channel chunks of 32; 3 WMMAs per chunk per tile.
// ---------------------------------------------------------------------------
extern __shared__ char hk_smem[];

__device__ __forceinline__ v16bf hk_cat(v8bf a, v8bf b) {
    return __builtin_shufflevector(a, b, 0,1,2,3,4,5,6,7,8,9,10,11,12,13,14,15);
}

__global__ __launch_bounds__(256) void hk_conv_wmma(
    const float*  __restrict__ x,     // (B, 64, 256, 256) f32
    const __bf16* __restrict__ Wt_hi, // (B, 64, 576) bf16, k = tap*64+ci
    const __bf16* __restrict__ Wt_lo, // (B, 64, 576) bf16
    const float*  __restrict__ bias,  // (B, 64) f32
    float* __restrict__ out)          // (B, 64, 256, 256) f32
{
    __bf16* patch_hi = (__bf16*)hk_smem;                       // 18*18*72
    __bf16* patch_lo = patch_hi + PCOLS * PCOLS * PSTR;

    const int blk = blockIdx.x;                 // B * 16 * 16 = 2048 blocks
    const int bx  = blk & 15;
    const int by  = (blk >> 4) & 15;
    const int b   = blk >> 8;
    const int x0  = bx << 4, y0 = by << 4;

    const int tid  = threadIdx.x;
    const int lane = tid & 31;
    const int wave = tid >> 5;

    // ---- cooperative halo-patch load + hi/lo split -------------------------
    const float* xb = x + (size_t)b * CIN * HW * HW;
    for (int p = wave; p < CIN * PCOLS; p += 8) {   // 1152 (ch,row) pairs
        int ch = p & 63;
        int ry = p >> 6;
        if (lane < PCOLS) {
            int gy = y0 - 1 + ry;
            int gx = x0 - 1 + lane;
            float v = 0.0f;
            if ((unsigned)gy < HW && (unsigned)gx < HW)
                v = xb[(size_t)ch * HW * HW + gy * HW + gx];
            __bf16 hi = (__bf16)v;
            __bf16 lo = (__bf16)(v - (float)hi);
            int idx = (ry * PCOLS + lane) * PSTR + ch;
            patch_hi[idx] = hi;
            patch_lo[idx] = lo;
        }
    }
    __syncthreads();

    // ---- per-wave GEMM tiles ----------------------------------------------
    const int otile  = wave & 3;                // out-ch group of 16
    const int pybase = (wave >> 2) << 3;        // rows 0..7 or 8..15
    const int m      = lane & 15;               // A: M index / B: N index
    const int hiLane = lane >> 4;               // 0 or 1
    const int koffA  = hiLane << 3;             // A K-offset: 0 or 8
    const int koffB  = hiLane << 3;             // B K-offset: 0 or 8

    const size_t arow = ((size_t)(b * COUT + otile * 16 + m)) * KDIM + koffA;
    const __bf16* Ah = Wt_hi + arow;
    const __bf16* Al = Wt_lo + arow;

    v8f acc[8];
#pragma unroll
    for (int j = 0; j < 8; ++j) acc[j] = (v8f){0,0,0,0,0,0,0,0};

    for (int tap = 0; tap < 9; ++tap) {
        const int dy = tap / 3, dx = tap % 3;
        const int colbase = m + dx;             // 0..17
#pragma unroll
        for (int chalf = 0; chalf < 2; ++chalf) {
            const int k0 = tap * CIN + chalf * 32;
            // A operand: K {koffA..+7, koffA+16..+23} -> 2 x b128, reused 8x
            v16bf ah = hk_cat(*(const v8bf*)(Ah + k0),
                              *(const v8bf*)(Ah + k0 + 16));
            v16bf al = hk_cat(*(const v8bf*)(Al + k0),
                              *(const v8bf*)(Al + k0 + 16));
#pragma unroll
            for (int j = 0; j < 8; ++j) {
                const int py = pybase + j;
                const int pidx = ((py + dy) * PCOLS + colbase) * PSTR
                               + chalf * 32 + koffB;
                v16bf bh = hk_cat(*(const v8bf*)(patch_hi + pidx),
                                  *(const v8bf*)(patch_hi + pidx + 16));
                v16bf bl = hk_cat(*(const v8bf*)(patch_lo + pidx),
                                  *(const v8bf*)(patch_lo + pidx + 16));
                acc[j] = __builtin_amdgcn_wmma_f32_16x16x32_bf16(
                    false, ah, false, bh, (short)0, acc[j], false, false);
                acc[j] = __builtin_amdgcn_wmma_f32_16x16x32_bf16(
                    false, ah, false, bl, (short)0, acc[j], false, false);
                acc[j] = __builtin_amdgcn_wmma_f32_16x16x32_bf16(
                    false, al, false, bh, (short)0, acc[j], false, false);
            }
        }
    }

    // ---- epilogue: bias + store -------------------------------------------
    // D layout: VGPR r, lanes0-15 -> M=r, lanes16-31 -> M=r+8; N = lane&15.
    const int ohalf = hiLane << 3;
    float bv[8];
#pragma unroll
    for (int r = 0; r < 8; ++r)
        bv[r] = bias[b * COUT + otile * 16 + r + ohalf];

#pragma unroll
    for (int j = 0; j < 8; ++j) {
        const int py = pybase + j;
#pragma unroll
        for (int r = 0; r < 8; ++r) {
            const int o = otile * 16 + r + ohalf;
            out[(((size_t)b * COUT + o) * HW + (y0 + py)) * HW + x0 + m]
                = acc[j][r] + bv[r];
        }
    }
}

// ---------------------------------------------------------------------------
extern "C" void kernel_launch(void* const* d_in, const int* in_sizes, int n_in,
                              void* d_out, int out_size, void* d_ws, size_t ws_size,
                              hipStream_t stream)
{
    const float* x           = (const float*)d_in[0];
    const float* h           = (const float*)d_in[1];
    const float* index       = (const float*)d_in[2];
    const float* conv_weight = (const float*)d_in[3];
    const float* conv_bias   = (const float*)d_in[4];
    const float* W_wp        = (const float*)d_in[5];
    const float* b_wp        = (const float*)d_in[6];
    const float* W_wi        = (const float*)d_in[7];
    const float* b_wi        = (const float*)d_in[8];
    const float* W_we        = (const float*)d_in[9];
    const float* b_we        = (const float*)d_in[10];
    const float* W_bp        = (const float*)d_in[11];
    const float* b_bp        = (const float*)d_in[12];
    const float* W_bi        = (const float*)d_in[13];
    const float* b_bi        = (const float*)d_in[14];
    float* out = (float*)d_out;

    // workspace layout: w1 (f32) | bias (f32) | Wt_hi (bf16) | Wt_lo (bf16)
    float*  w1    = (float*)d_ws;                           // 32768 f32
    float*  biasb = w1 + BATCH * CIN * MID;                 // 512 f32
    __bf16* Wt_hi = (__bf16*)(biasb + BATCH * COUT);        // 294912 bf16
    __bf16* Wt_lo = Wt_hi + (size_t)BATCH * COUT * KDIM;    // 294912 bf16

    hipLaunchKernelGGL(hk_embed_w1, dim3(128), dim3(256), 0, stream,
                       h, index, W_wp, b_wp, W_wi, b_wi, w1);
    hipLaunchKernelGGL(hk_embed_w2, dim3(1152), dim3(256), 0, stream,
                       w1, W_we, b_we, conv_weight, Wt_hi, Wt_lo);
    hipLaunchKernelGGL(hk_embed_bias, dim3(2), dim3(256), 0, stream,
                       h, index, W_bp, b_bp, W_bi, b_bi, conv_bias, biasb);

    const size_t lds_bytes =
        (size_t)2 * PCOLS * PCOLS * PSTR * sizeof(__bf16);  // 93312 B
    hipLaunchKernelGGL(hk_conv_wmma, dim3(BATCH * 16 * 16), dim3(256),
                       lds_bytes, stream, x, Wt_hi, Wt_lo, biasb, out);
}
